// ClusterModel_23098334118129
// MI455X (gfx1250) — matrix operand order repaired and checked by reading, not compiled, not run on hardware
//
#include <hip/hip_runtime.h>
#include <hip/hip_bf16.h>
#include <math.h>

typedef __bf16 bf16_t;
typedef __attribute__((ext_vector_type(16))) __bf16 v16bf;
typedef __attribute__((ext_vector_type(8)))  float  v8f;
typedef __attribute__((ext_vector_type(4))) unsigned int u32x4;
typedef __attribute__((ext_vector_type(8))) unsigned int u32x8;

#define CIN_DIM 128
#define H_DIM   64
#define K_DIM   100
#define KP      112   // K padded to 7*16
#define SS_CHUNK 1024

// LDS row paddings (+8 bf16 = 16B keeps 16B alignment, breaks bank conflicts)
#define CIN_P (CIN_DIM + 8)   // 272B stride -> banks 4L%64 distinct
#define H_P   (H_DIM + 8)     // 144B stride -> banks 36L%64 distinct
#define SSK_P (32 + 8)        //  80B stride -> banks 20L%64 distinct

// ---------- wave32 reductions ----------
static __device__ __forceinline__ float wave_sum(float v) {
#pragma unroll
  for (int off = 16; off > 0; off >>= 1) v += __shfl_xor(v, off, 32);
  return v;
}
static __device__ __forceinline__ float wave_max(float v) {
#pragma unroll
  for (int off = 16; off > 0; off >>= 1) v = fmaxf(v, __shfl_xor(v, off, 32));
  return v;
}

// Build v16bf operands from contiguous LDS runs (vectorizes to 2x ds_load_b128)
static __device__ __forceinline__ v16bf load_a(const bf16_t* row, int kbA) {
  v16bf a;
  const bf16_t* p0 = row + kbA;        // K = kbA .. kbA+7
  const bf16_t* p1 = row + 16 + kbA;   // K = 16+kbA .. 16+kbA+7
#pragma unroll
  for (int e = 0; e < 8; ++e) { a[e] = p0[e]; a[e + 8] = p1[e]; }
  return a;
}
static __device__ __forceinline__ v16bf load_b(const bf16_t* row, int kbB) {
  v16bf b;
  const bf16_t* p0 = row + kbB;        // K = kbB .. kbB+7
  const bf16_t* p1 = row + 16 + kbB;   // K = 16+kbB .. 16+kbB+7
#pragma unroll
  for (int e = 0; e < 8; ++e) { b[e] = p0[e]; b[e + 8] = p1[e]; }
  return b;
}

// ---------- utility ----------
__global__ void k_zero(float* __restrict__ p, int n) {
  int i = blockIdx.x * blockDim.x + threadIdx.x;
  if (i < n) p[i] = 0.0f;
}

// ---------- GEMM1: hx = x @ W1   (N x 128) @ (128 x 64), bf16 WMMA ----------
__global__ void k_gemm1(const float* __restrict__ x, const float* __restrict__ W1,
                        float* __restrict__ hx, int N) {
  __shared__ __align__(16) bf16_t wldsT[H_DIM][CIN_P];  // transposed: [col][k]
  __shared__ __align__(16) bf16_t xlds[32][CIN_P];      // [row][k]
  const int tid = threadIdx.x;
  for (int i = tid; i < H_DIM * CIN_DIM; i += 256) {
    int col = i >> 7;            // / CIN_DIM
    int k   = i & (CIN_DIM - 1);
    wldsT[col][k] = (bf16_t)W1[(long)k * H_DIM + col];
  }
  const int rowBase = blockIdx.x * 32;
  for (int i = tid; i < 32 * (CIN_DIM / 4); i += 256) {
    int r = i >> 5;
    int c4 = i & 31;
    int gr = rowBase + r;
    float4 v = make_float4(0.f, 0.f, 0.f, 0.f);
    if (gr < N) v = ((const float4*)x)[((long)gr * CIN_DIM >> 2) + c4];
    bf16_t* d = &xlds[r][c4 * 4];
    d[0] = (bf16_t)v.x; d[1] = (bf16_t)v.y; d[2] = (bf16_t)v.z; d[3] = (bf16_t)v.w;
  }
  __syncthreads();

  const int wave = tid >> 5, lane = tid & 31;
  const int rt = wave >> 2;        // row tile 0..1
  const int ct = wave & 3;         // col tile 0..3
  const int lm = lane & 15, lh = lane >> 4;
  const int kbA = lh * 8, kbB = lh * 8;
  const bf16_t* arow = &xlds[rt * 16 + lm][0];
  const bf16_t* brow = &wldsT[ct * 16 + lm][0];
  v8f acc = {};
#pragma unroll
  for (int kk = 0; kk < CIN_DIM; kk += 32) {
    v16bf a = load_a(arow + kk, kbA);
    v16bf b = load_b(brow + kk, kbB);
    acc = __builtin_amdgcn_wmma_f32_16x16x32_bf16(false, a, false, b,
                                                  (short)0, acc, false, false);
  }
  // hoisted (wave-uniform) bounds: fast path has zero per-store guards
  const int col = ct * 16 + lm;
  float* out0 = hx + (long)(rowBase + rt * 16 + lh * 8) * H_DIM + col;
  if (rowBase + 32 <= N) {
#pragma unroll
    for (int r8 = 0; r8 < 8; ++r8) out0[(long)r8 * H_DIM] = acc[r8];
  } else {
#pragma unroll
    for (int r8 = 0; r8 < 8; ++r8) {
      int row = rowBase + rt * 16 + r8 + lh * 8;
      if (row < N) hx[(long)row * H_DIM + col] = acc[r8];
    }
  }
}

// ---------- degree / normalization ----------
__global__ void k_deg_scatter(const int* __restrict__ dst, const float* __restrict__ w,
                              float* __restrict__ deg_nl, int E) {
  int e = blockIdx.x * blockDim.x + threadIdx.x;
  if (e < E) atomicAdd(&deg_nl[dst[e]], w[e]);
}

__global__ void k_inv(const float* __restrict__ deg_nl, float* __restrict__ inv, int N) {
  int n = blockIdx.x * blockDim.x + threadIdx.x;
  if (n < N) {
    float d = deg_nl[n] + 1.0f;     // + self loop weight 1
    inv[n] = (d > 0.0f) ? rsqrtf(d) : 0.0f;
  }
}

__global__ void k_selfloop(const float* __restrict__ hx, const float* __restrict__ inv,
                           float* __restrict__ agg, int N) {
  int i = blockIdx.x * blockDim.x + threadIdx.x;
  if (i < N * H_DIM) {
    float iv = inv[i >> 6];
    agg[i] = iv * iv * hx[i];
  }
}

// one wave per edge; lane c handles channels c and c+32
__global__ void k_edge_agg(const int* __restrict__ src, const int* __restrict__ dst,
                           const float* __restrict__ w, const float* __restrict__ inv,
                           const float* __restrict__ hx, float* __restrict__ agg, int E) {
  int wid = (blockIdx.x * blockDim.x + threadIdx.x) >> 5;
  int lane = threadIdx.x & 31;
  if (wid >= E) return;
  int s = src[wid], d = dst[wid];
  float coef = inv[s] * inv[d] * w[wid];
  const float* hs = hx + (long)s * H_DIM;
  float* ad = agg + (long)d * H_DIM;
  atomicAdd(&ad[lane],      coef * hs[lane]);
  atomicAdd(&ad[lane + 32], coef * hs[lane + 32]);
}

__global__ void k_elu_bias(const float* __restrict__ agg, const float* __restrict__ b1,
                           float* __restrict__ h, int N) {
  int i = blockIdx.x * blockDim.x + threadIdx.x;
  if (i < N * H_DIM) {
    float v = agg[i] + b1[i & (H_DIM - 1)];
    h[i] = (v > 0.0f) ? v : (expf(v) - 1.0f);   // ELU
  }
}

// ---------- GEMM2: logits = h @ Wp + bp  -> padded [N x KP] ----------
__global__ void k_gemm2(const float* __restrict__ h, const float* __restrict__ Wp,
                        const float* __restrict__ bp, float* __restrict__ logits, int N) {
  __shared__ __align__(16) bf16_t wldsT[KP][H_P];    // transposed: [col][k]
  __shared__ __align__(16) bf16_t hlds[16][H_P];     // [row][k]
  const int tid = threadIdx.x;
  for (int i = tid; i < KP * H_DIM; i += 256) {
    int col = i >> 6;            // / H_DIM
    int k   = i & (H_DIM - 1);
    wldsT[col][k] = (col < K_DIM) ? (bf16_t)Wp[(long)k * K_DIM + col] : (bf16_t)0.0f;
  }
  const int rowBase = blockIdx.x * 16;
  for (int i = tid; i < 16 * H_DIM; i += 256) {
    int r = i >> 6, c = i & (H_DIM - 1);
    int gr = rowBase + r;
    hlds[r][c] = (gr < N) ? (bf16_t)h[(long)gr * H_DIM + c] : (bf16_t)0.0f;
  }
  __syncthreads();

  const int wave = tid >> 5, lane = tid & 31;
  if (wave >= 7) return;               // 7 column tiles cover KP=112
  const int lm = lane & 15, lh = lane >> 4;
  const int kbA = lh * 8, kbB = lh * 8;
  const bf16_t* arow = &hlds[lm][0];
  const bf16_t* brow = &wldsT[wave * 16 + lm][0];
  v8f acc = {};
#pragma unroll
  for (int kk = 0; kk < H_DIM; kk += 32) {
    v16bf a = load_a(arow + kk, kbA);
    v16bf b = load_b(brow + kk, kbB);
    acc = __builtin_amdgcn_wmma_f32_16x16x32_bf16(false, a, false, b,
                                                  (short)0, acc, false, false);
  }
  const int col = wave * 16 + lm;                       // < KP (padded store)
  const float bias = (col < K_DIM) ? bp[col] : 0.0f;    // guard only the read
  float* out0 = logits + (long)(rowBase + lh * 8) * KP + col;
  if (rowBase + 16 <= N) {
#pragma unroll
    for (int r8 = 0; r8 < 8; ++r8) out0[(long)r8 * KP] = acc[r8] + bias;
  } else {
#pragma unroll
    for (int r8 = 0; r8 < 8; ++r8) {
      int row = rowBase + r8 + lh * 8;
      if (row < N) logits[(long)row * KP + col] = acc[r8] + bias;
    }
  }
}

// ---------- row softmax over 100 of each padded KP row -> s into d_out ----------
__global__ void k_softmax(const float* __restrict__ logits, float* __restrict__ s_out, int N) {
  int wid = (blockIdx.x * blockDim.x + threadIdx.x) >> 5;
  int lane = threadIdx.x & 31;
  if (wid >= N) return;
  const float* row = logits + (long)wid * KP;
  float v0 = row[lane], v1 = row[lane + 32];
  float v2 = (lane + 64 < K_DIM) ? row[lane + 64] : -INFINITY;
  float m = wave_max(fmaxf(fmaxf(v0, v1), v2));
  float e0 = expf(v0 - m), e1 = expf(v1 - m);
  float e2 = (lane + 64 < K_DIM) ? expf(v2 - m) : 0.0f;
  float rs = 1.0f / wave_sum(e0 + e1 + e2);
  float* out = s_out + (long)wid * K_DIM;
  out[lane] = e0 * rs;
  out[lane + 32] = e1 * rs;
  if (lane + 64 < K_DIM) out[lane + 64] = e2 * rs;
}

// ---------- cut numerator: sum_e w_e * <s[src], s[dst]> ----------
__global__ void k_cut_num(const int* __restrict__ src, const int* __restrict__ dst,
                          const float* __restrict__ w, const float* __restrict__ s,
                          float* __restrict__ num_acc, int E) {
  __shared__ float blk;
  if (threadIdx.x == 0) blk = 0.0f;
  __syncthreads();
  int wid = (blockIdx.x * blockDim.x + threadIdx.x) >> 5;
  int lane = threadIdx.x & 31;
  if (wid < E) {
    const float* ss = s + (long)src[wid] * K_DIM;
    const float* sd = s + (long)dst[wid] * K_DIM;
    __builtin_prefetch(ss, 0, 0);                   // global_prefetch_b8
    float dot = ss[lane] * sd[lane] + ss[lane + 32] * sd[lane + 32];
    if (lane + 64 < K_DIM) dot += ss[lane + 64] * sd[lane + 64];
    dot = wave_sum(dot);
    if (lane == 0) atomicAdd(&blk, w[wid] * dot);
  }
  __syncthreads();
  if (threadIdx.x == 0) atomicAdd(num_acc, blk);
}

// ---------- denominator: sum_n deg_nl[n] * <s[n], s[n]> ----------
__global__ void k_den(const float* __restrict__ s, const float* __restrict__ deg_nl,
                      float* __restrict__ den_acc, int N) {
  __shared__ float blk;
  if (threadIdx.x == 0) blk = 0.0f;
  __syncthreads();
  int wid = (blockIdx.x * blockDim.x + threadIdx.x) >> 5;
  int lane = threadIdx.x & 31;
  if (wid < N) {
    const float* sr = s + (long)wid * K_DIM;
    float d = sr[lane] * sr[lane] + sr[lane + 32] * sr[lane + 32];
    if (lane + 64 < K_DIM) d += sr[lane + 64] * sr[lane + 64];
    d = wave_sum(d);
    if (lane == 0) atomicAdd(&blk, deg_nl[wid] * d);
  }
  __syncthreads();
  if (threadIdx.x == 0) atomicAdd(den_acc, blk);
}

// ---------- SS = s^T @ s via WMMA, staged by the Tensor Data Mover ----------
// Per 32-row chunk: wave 0 issues a TDM 2D tile load (100 x 32 f32, OOB rows
// return zero) into LDS, waits on TENSORcnt, then all waves transpose+convert
// to bf16 sldsT[col][n] so WMMA operands stay contiguous ds_load_b128 runs.
__global__ void k_ss(const float* __restrict__ s, float* __restrict__ SS, int N) {
  __shared__ __align__(16) float  sf32[32 * K_DIM];    // 12.8 KB TDM landing pad
  __shared__ __align__(16) bf16_t sldsT[KP][SSK_P];    // ~9 KB operand store
  const int tid = threadIdx.x, wave = tid >> 5, lane = tid & 31;
  const int lm = lane & 15, lh = lane >> 4;
  const int kbA = lh * 8, kbB = lh * 8;
  v8f zero = {};
  v8f acc[7];
#pragma unroll
  for (int i = 0; i < 7; ++i) acc[i] = zero;

  const unsigned lds_off = (unsigned)(size_t)&sf32[0];
  const int base = blockIdx.x * SS_CHUNK;
  for (int n0 = base; n0 < base + SS_CHUNK; n0 += 32) {
    __syncthreads();                       // protect prior iteration's reads
    if (wave == 0) {
      unsigned long long ga = (unsigned long long)(size_t)(s + (size_t)n0 * K_DIM);
      int rows = N - n0; if (rows < 0) rows = 0;   // OOB rows -> TDM zero-fill
      u32x4 g0;
      g0[0] = 1u;                                   // count=1 (valid descriptor)
      g0[1] = lds_off;                              // lds_addr
      g0[2] = (unsigned)ga;                         // global_addr[31:0]
      g0[3] = (unsigned)(ga >> 32) | (2u << 30);    // global_addr[56:32] | type=2
      u32x8 g1;
      g1[0] = 2u << 16;                             // data_size=2 (4 bytes)
      g1[1] = (unsigned)K_DIM << 16;                // tensor_dim0 = 100
      g1[2] = ((unsigned)rows & 0xFFFFu) << 16;     // tensor_dim1 lo
      g1[3] = (((unsigned)rows >> 16) & 0xFFFFu) | ((unsigned)K_DIM << 16); // dim1 hi | tile_dim0=100
      g1[4] = 32u;                                  // tile_dim1 = 32 rows
      g1[5] = (unsigned)K_DIM;                      // tensor_dim0_stride = 100
      g1[6] = 0u;                                   // stride hi | dim1_stride lo
      g1[7] = 0u;
      asm volatile("tensor_load_to_lds %0, %1" :: "s"(g0), "s"(g1) : "memory");
      __builtin_amdgcn_s_wait_tensorcnt(0);
    }
    __syncthreads();                       // TDM data visible to all waves
    // transpose + convert: sldsT[c][r] = (bf16) sf32[r][c], zero-pad c>=100
    for (int i = tid; i < 32 * KP; i += 256) {
      int r = i / KP, c = i - r * KP;
      sldsT[c][r] = (c < K_DIM) ? (bf16_t)sf32[r * K_DIM + c] : (bf16_t)0.0f;
    }
    __syncthreads();
#pragma unroll
    for (int idx = 0; idx < 7; ++idx) {
      int t = wave + idx * 8;
      if (t >= 49) break;                  // wave-uniform: EXEC stays all-ones
      int ti = t / 7, tj = t % 7;
      v16bf a = load_a(&sldsT[ti * 16 + lm][0], kbA);   // A[m][k] = s[k][m]
      v16bf b = load_b(&sldsT[tj * 16 + lm][0], kbB);   // B[k][n] = s[k][n]
      acc[idx] = __builtin_amdgcn_wmma_f32_16x16x32_bf16(false, a, false, b,
                                                         (short)0, acc[idx], false, false);
    }
  }
#pragma unroll
  for (int idx = 0; idx < 7; ++idx) {
    int t = wave + idx * 8;
    if (t >= 49) break;
    int ti = t / 7, tj = t % 7;
#pragma unroll
    for (int r8 = 0; r8 < 8; ++r8) {
      int m = ti * 16 + r8 + lh * 8;
      int n = tj * 16 + lm;
      atomicAdd(&SS[m * KP + n], acc[idx][r8]);
    }
  }
}

// ---------- finalize losses ----------
__global__ void k_final(const float* __restrict__ SS, const float* __restrict__ num,
                        const float* __restrict__ den, float* __restrict__ out_scal) {
  __shared__ float red[256];
  __shared__ float fro_s;
  const int tid = threadIdx.x;
  float acc = 0.0f;
  for (int i = tid; i < KP * KP; i += 256) { float v = SS[i]; acc += v * v; }
  red[tid] = acc; __syncthreads();
  for (int s2 = 128; s2 > 0; s2 >>= 1) {
    if (tid < s2) red[tid] += red[tid + s2];
    __syncthreads();
  }
  if (tid == 0) fro_s = sqrtf(red[0]);
  __syncthreads();
  float inv_fro = (fro_s > 0.0f) ? (1.0f / fro_s) : 0.0f;
  const float invsqK = rsqrtf((float)K_DIM);
  acc = 0.0f;
  for (int i = tid; i < K_DIM * K_DIM; i += 256) {
    int r = i / K_DIM, c = i % K_DIM;
    float v = SS[r * KP + c] * inv_fro - ((r == c) ? invsqK : 0.0f);
    acc += v * v;
  }
  red[tid] = acc; __syncthreads();
  for (int s2 = 128; s2 > 0; s2 >>= 1) {
    if (tid < s2) red[tid] += red[tid + s2];
    __syncthreads();
  }
  if (tid == 0) {
    float ortho = sqrtf(red[0]);
    float cut = -(*num) / (*den);
    out_scal[0] = cut + ortho;   // aux_loss
    out_scal[1] = cut;           // cut_loss
    out_scal[2] = ortho;         // ortho_loss
  }
}

// =====================================================================
extern "C" void kernel_launch(void* const* d_in, const int* in_sizes, int n_in,
                              void* d_out, int out_size, void* d_ws, size_t ws_size,
                              hipStream_t stream) {
  const float* x  = (const float*)d_in[0];
  const float* W1 = (const float*)d_in[1];
  const float* b1 = (const float*)d_in[2];
  const float* Wp = (const float*)d_in[3];
  const float* bp = (const float*)d_in[4];
  const int*   ei = (const int*)d_in[5];
  const float* ew = (const float*)d_in[6];
  // d_in[7] = batch (single graph) unused

  const int N = in_sizes[0] / CIN_DIM;
  const int E = in_sizes[5] / 2;
  const int* src = ei;
  const int* dst = ei + E;

  char* ws = (char*)d_ws;
  size_t off = 0;
  float* hx   = (float*)(ws + off); off += (size_t)N * H_DIM * 4;   // x@W1, later h
  float* agg  = (float*)(ws + off); off += (size_t)N * H_DIM * 4;   // aggregation
  float* inv  = (float*)(ws + off); off += (size_t)N * 4;
  float* degn = (float*)(ws + off); off += (size_t)N * 4;
  float* logi = (float*)(ws + off); off += (size_t)N * KP * 4;      // padded stride
  float* SS   = (float*)(ws + off); off += (size_t)KP * KP * 4;
  float* scal = (float*)(ws + off); off += 16;                      // [num, den]

  float* s_out    = (float*)d_out;                 // [N*K]
  float* out_scal = s_out + (size_t)N * K_DIM;     // 3 scalars

  const dim3 blk(256);
  const unsigned EW = (unsigned)(((size_t)E * 32 + 255) / 256);   // wave-per-edge grids
  const unsigned NW = (unsigned)(((size_t)N * 32 + 255) / 256);   // wave-per-node grids
  const unsigned NC = (unsigned)(((size_t)N * H_DIM + 255) / 256);

  k_zero<<<(N + 255) / 256, blk, 0, stream>>>(degn, N);
  k_zero<<<(KP * KP + 255) / 256, blk, 0, stream>>>(SS, KP * KP);
  k_zero<<<1, blk, 0, stream>>>(scal, 4);

  k_gemm1<<<(N + 31) / 32, blk, 0, stream>>>(x, W1, hx, N);
  k_deg_scatter<<<(E + 255) / 256, blk, 0, stream>>>(dst, ew, degn, E);
  k_inv<<<(N + 255) / 256, blk, 0, stream>>>(degn, inv, N);
  k_selfloop<<<NC, blk, 0, stream>>>(hx, inv, agg, N);
  k_edge_agg<<<EW, blk, 0, stream>>>(src, dst, ew, inv, hx, agg, E);
  k_elu_bias<<<NC, blk, 0, stream>>>(agg, b1, hx, N);              // h -> hx (in place reuse)
  k_gemm2<<<(N + 15) / 16, blk, 0, stream>>>(hx, Wp, bp, logi, N);
  k_softmax<<<NW, blk, 0, stream>>>(logi, s_out, N);
  k_cut_num<<<EW, blk, 0, stream>>>(src, dst, ew, s_out, &scal[0], E);
  k_den<<<NW, blk, 0, stream>>>(s_out, degn, &scal[1], N);
  k_ss<<<(N + SS_CHUNK - 1) / SS_CHUNK, blk, 0, stream>>>(s_out, SS, N);
  k_final<<<1, blk, 0, stream>>>(SS, &scal[0], &scal[1], out_scal);
}
